// DiTBlock_63410897158587
// MI455X (gfx1250) — compile-verified, matrix operand order checked by reading, not run on hardware
//
#include <hip/hip_runtime.h>
#include <hip/hip_bf16.h>
#include <math.h>
#include <stdint.h>

// DiT block for MI455X (gfx1250): bf16 WMMA GEMMs (TDM-staged LDS tiles,
// double-buffered) + flash attention with DPP16 softmax reductions.

#define H_    1152
#define B_    16
#define N_    1024
#define NH_   16
#define HD_   72
#define HDP_  96          // head dim padded to 3*32 for WMMA K
#define MLP_  4608
#define MOD6_ (6 * H_)
#define ROWS_ (B_ * N_)   // 16384
#define BM_   128         // GEMM block rows (8 waves x 16)
#define BK_   64          // GEMM K tile

typedef __attribute__((ext_vector_type(16))) __bf16 v16bf;
typedef __attribute__((ext_vector_type(8)))  __bf16 v8bf;
typedef __attribute__((ext_vector_type(8)))  float  v8f;
typedef __attribute__((ext_vector_type(4)))  unsigned int v4u;
typedef __attribute__((ext_vector_type(8)))  unsigned int v8u;

static __device__ __forceinline__ v8f wmma_bf16(v16bf a, v16bf b, v8f c) {
  return __builtin_amdgcn_wmma_f32_16x16x32_bf16(false, a, false, b, (short)0, c,
                                                 false, false);
}

static __device__ __forceinline__ v8f zero8() {
  v8f z;
#pragma unroll
  for (int i = 0; i < 8; ++i) z[i] = 0.0f;
  return z;
}

// 16x32 bf16 fragment (A layout; B mirrored with row = output column,
// source [N][K]). Lane L: row = row0+(L&15); elems 0..7 = K kb..kb+7,
// 8..15 = K kb+16..kb+23, kb = k0 + 8*(L>>4). Two 16B loads per lane.
static __device__ __forceinline__ v16bf load_frag(const __bf16* __restrict__ base,
                                                  int ld, int row0, int k0, int lane) {
  const int r  = row0 + (lane & 15);
  const int kb = k0 + ((lane >> 4) << 3);
  const __bf16* p = base + (size_t)r * ld + kb;
  v8bf lo = *(const v8bf*)p;
  v8bf hi = *(const v8bf*)(p + 16);
  v16bf f;
#pragma unroll
  for (int i = 0; i < 8; ++i) { f[i] = lo[i]; f[i + 8] = hi[i]; }
  return f;
}

// ---------------- DPP16 butterfly reductions over 16-lane halves (pure VALU)
#define DPPF_(x, ctrl) \
  __int_as_float(__builtin_amdgcn_update_dpp(0, __float_as_int(x), (ctrl), 0xF, 0xF, true))

static __device__ __forceinline__ float rowmax16(float v) {
  v = fmaxf(v, DPPF_(v, 0xB1));   // quad_perm [1,0,3,2]  (xor 1)
  v = fmaxf(v, DPPF_(v, 0x4E));   // quad_perm [2,3,0,1]  (xor 2)
  v = fmaxf(v, DPPF_(v, 0x141));  // row_half_mirror      (pairs 0-7 / 7-0)
  v = fmaxf(v, DPPF_(v, 0x140));  // row_mirror           (pairs 0-15 / 15-0)
  return v;
}
static __device__ __forceinline__ float rowsum16(float v) {
  v += DPPF_(v, 0xB1);
  v += DPPF_(v, 0x4E);
  v += DPPF_(v, 0x141);
  v += DPPF_(v, 0x140);
  return v;
}

// ----------------------------------- Tensor Data Mover: 2D tile -> LDS (async)
// D# group0: [1:0]=count=1, [63:32]=lds_addr, [120:64]=global_addr, [127:126]=type(2)
// D# group1: data_size=1(2B), tensor_dim0=k_len, tensor_dim1=2^20,
//            tile_dim0=tile_k, tile_dim1=rows, tensor_dim0_stride=stride
static __device__ __forceinline__ void tdm_load_2d(__bf16* lds_dst, const __bf16* gsrc,
                                                   unsigned k_len, unsigned tile_k,
                                                   unsigned rows, unsigned stride) {
  const unsigned long long ga = (unsigned long long)(uintptr_t)gsrc;
  const unsigned lds = (unsigned)(uintptr_t)lds_dst;
  const unsigned dim1 = 1u << 20;
  v4u g0;
  g0[0] = 1u;                                   // count=1 (user descriptor)
  g0[1] = lds;                                  // lds_addr
  g0[2] = (unsigned)ga;                         // global_addr[31:0]
  g0[3] = (unsigned)(ga >> 32) | (2u << 30);    // global_addr[56:32] | type=2
  v8u g1;
  g1[0] = (1u << 16);                           // data_size=1 -> 2 bytes/elem
  g1[1] = (k_len & 0xFFFFu) << 16;              // tensor_dim0[15:0]
  g1[2] = (k_len >> 16) | ((dim1 & 0xFFFFu) << 16);
  g1[3] = (dim1 >> 16) | (tile_k << 16);        // tensor_dim1[31:16] | tile_dim0
  g1[4] = rows;                                 // tile_dim1 (tile_dim2 = 0)
  g1[5] = stride;                               // tensor_dim0_stride[31:0]
  g1[6] = 0;                                    // stride[47:32] | dim1_stride lo
  g1[7] = 0;
  asm volatile("tensor_load_to_lds %0, %1" :: "s"(g0), "s"(g1) : "memory");
}

// ---------------------------------------------------------------- small prep
__global__ void swish_kernel(const float* __restrict__ c, float* __restrict__ out, int n) {
  int i = blockIdx.x * 256 + threadIdx.x;
  if (i < n) { float v = c[i]; out[i] = v / (1.0f + __expf(-v)); }
}

__global__ void mod_gemm_kernel(const float* __restrict__ sc, const float* __restrict__ w,
                                const float* __restrict__ bvec, float* __restrict__ out) {
  int idx = blockIdx.x * 256 + threadIdx.x;
  if (idx >= B_ * MOD6_) return;
  int b = idx / MOD6_, col = idx - b * MOD6_;
  float acc = bvec[col];
  const float* s = sc + b * H_;
  for (int k = 0; k < H_; ++k) acc = fmaf(s[k], w[(size_t)k * MOD6_ + col], acc);
  out[idx] = acc;
}

// fp32 [K][N] -> bf16 [N][K]
__global__ void convT_kernel(const float* __restrict__ in, __bf16* __restrict__ out,
                             int K, int N) {
  int idx = blockIdx.x * 256 + threadIdx.x;
  if (idx >= K * N) return;
  int k = idx / N, n = idx - k * N;
  out[(size_t)n * K + k] = (__bf16)in[idx];
}

// -------------------------------------------------- LayerNorm + adaLN modulate
__global__ void ln_mod_kernel(const float* __restrict__ x, const float* __restrict__ mod,
                              int sh_off, int sc_off, __bf16* __restrict__ out) {
  __shared__ float r1[256], r2[256];
  const int row = blockIdx.x;
  const int b = row >> 10;
  const int t = threadIdx.x;
  const float* xr = x + (size_t)row * H_;
  float s = 0.0f, s2 = 0.0f;
  for (int i = t; i < H_; i += 256) { float v = xr[i]; s += v; s2 += v * v; }
  r1[t] = s; r2[t] = s2;
  __syncthreads();
  for (int off = 128; off > 0; off >>= 1) {
    if (t < off) { r1[t] += r1[t + off]; r2[t] += r2[t + off]; }
    __syncthreads();
  }
  const float mean = r1[0] * (1.0f / H_);
  const float var  = r2[0] * (1.0f / H_) - mean * mean;
  const float rstd = rsqrtf(var + 1e-6f);
  const float* shp = mod + (size_t)b * MOD6_ + sh_off;
  const float* scp = mod + (size_t)b * MOD6_ + sc_off;
  __bf16* o = out + (size_t)row * H_;
  for (int i = t; i < H_; i += 256) {
    float y = (xr[i] - mean) * rstd;
    o[i] = (__bf16)(y * (1.0f + scp[i]) + shp[i]);
  }
}

// ------------------------------------------------------------ WMMA GEMM core
// C[MxN] = A[MxK](bf16) * Bt[NxK](bf16)^T + bias.
// Block: 8 waves; BM=128 rows (16 per wave), 64 cols, K staged in 64-wide
// tiles through LDS by the Tensor Data Mover, double buffered.
// EPI: 0 = bias->bf16 ; 1 = bias+GELU->bf16 ; 2 = resid + gate*(v+bias)->f32
template <int EPI>
__global__ void __launch_bounds__(256) gemm_kernel(
    const __bf16* __restrict__ A, const __bf16* __restrict__ Bt,
    const float* __restrict__ bias, float* __restrict__ Cf, __bf16* __restrict__ Cb,
    const float* __restrict__ resid, const float* __restrict__ gate,
    int M, int N, int K) {
  __shared__ __bf16 sA[2][BM_ * BK_];   // 2 x 16 KB
  __shared__ __bf16 sB[2][64 * BK_];    // 2 x  8 KB
  const int lane = threadIdx.x & 31;
  const int wave = threadIdx.x >> 5;
  const int tilesN = N >> 6;
  const int bm = blockIdx.x / tilesN, bn = blockIdx.x - bm * tilesN;
  const int row0 = bm * BM_, col0 = bn << 6;
  const int nk = K >> 6;   // 64-wide K stages

  v8f acc[4];
#pragma unroll
  for (int t = 0; t < 4; ++t) acc[t] = zero8();

  if (wave == 0) {
    tdm_load_2d(&sA[0][0], A + (size_t)row0 * K, (unsigned)K, BK_, BM_, (unsigned)K);
    tdm_load_2d(&sB[0][0], Bt + (size_t)col0 * K, (unsigned)K, BK_, 64, (unsigned)K);
  }

  for (int kb = 0; kb < nk; ++kb) {
    const int cur = kb & 1;
    if (wave == 0) {
      if (kb + 1 < nk) {
        const int k0n = (kb + 1) << 6;
        tdm_load_2d(&sA[cur ^ 1][0], A + (size_t)row0 * K + k0n, (unsigned)K, BK_, BM_,
                    (unsigned)K);
        tdm_load_2d(&sB[cur ^ 1][0], Bt + (size_t)col0 * K + k0n, (unsigned)K, BK_, 64,
                    (unsigned)K);
        __builtin_amdgcn_s_wait_tensorcnt(2);   // current stage done; next in flight
      } else {
        __builtin_amdgcn_s_wait_tensorcnt(0);
      }
    }
    __syncthreads();
    const __bf16* a_s = &sA[cur][0];
    const __bf16* b_s = &sB[cur][0];
#pragma unroll
    for (int kk = 0; kk < BK_; kk += 32) {
      v16bf a = load_frag(a_s, BK_, wave * 16, kk, lane);
#pragma unroll
      for (int t = 0; t < 4; ++t) {
        v16bf bf = load_frag(b_s, BK_, t * 16, kk, lane);
        acc[t] = wmma_bf16(a, bf, acc[t]);
      }
    }
    __syncthreads();   // everyone done reading before TDM overwrites this buffer
  }

  const int col = lane & 15, half = lane >> 4;
#pragma unroll
  for (int t = 0; t < 4; ++t) {
#pragma unroll
    for (int i = 0; i < 8; ++i) {
      const int m = row0 + wave * 16 + i + (half << 3);
      const int n = col0 + t * 16 + col;
      float v = acc[t][i] + bias[n];
      const size_t o = (size_t)m * N + n;
      if (EPI == 0) {
        Cb[o] = (__bf16)v;
      } else if (EPI == 1) {
        float g = 0.5f * v * (1.0f + tanhf(0.7978845608f * (v + 0.044715f * v * v * v)));
        Cb[o] = (__bf16)g;
      } else {
        const int bb = m >> 10;
        Cf[o] = resid[o] + gate[(size_t)bb * MOD6_ + n] * v;
      }
    }
  }
}

// ------------------------- per-head q/k LayerNorm + pad; V transposed (d-major)
__global__ void qk_ln_pack_kernel(const __bf16* __restrict__ qkv,
                                  const float* __restrict__ gq, const float* __restrict__ gk,
                                  __bf16* __restrict__ qp, __bf16* __restrict__ kp,
                                  __bf16* __restrict__ vT) {
  const int idx = blockIdx.x * 256 + threadIdx.x;       // (b*16+h)*1024 + n
  const int bh = idx >> 10, n = idx & 1023;
  const __bf16* base = qkv + (size_t)((bh >> 4) * N_ + n) * (3 * H_) + (bh & 15) * HD_;
#pragma unroll
  for (int s = 0; s < 2; ++s) {
    const __bf16* src = base + s * H_;
    const float* g = s ? gk : gq;
    __bf16* dst = (s ? kp : qp) + ((size_t)bh * N_ + n) * HDP_;
    float su = 0.0f, sq = 0.0f;
    for (int d = 0; d < HD_; ++d) { float v = (float)src[d]; su += v; sq += v * v; }
    float mean = su * (1.0f / HD_);
    float rstd = rsqrtf(sq * (1.0f / HD_) - mean * mean + 1e-6f);
    for (int d = 0; d < HDP_; ++d)
      dst[d] = (d < HD_) ? (__bf16)(((float)src[d] - mean) * rstd * g[d]) : (__bf16)0.0f;
  }
  const __bf16* vsrc = base + 2 * H_;
  for (int d = 0; d < HDP_; ++d)
    vT[((size_t)bh * HDP_ + d) * N_ + n] = (d < HD_) ? vsrc[d] : (__bf16)0.0f;
}

// ------------------------------------------------------------ flash attention
__global__ void __launch_bounds__(256) attn_kernel(const __bf16* __restrict__ qp,
                                                   const __bf16* __restrict__ kp,
                                                   const __bf16* __restrict__ vT,
                                                   __bf16* __restrict__ o) {
  __shared__ __bf16 sP[8][16 * 32];
  const int lane = threadIdx.x & 31;
  const int wave = threadIdx.x >> 5;
  const int task = blockIdx.x * 8 + wave;      // (b*16+h)*64 + qt
  const int qt = task & 63, bh = task >> 6;
  const int b = bh >> 4, h = bh & 15;
  const __bf16* qb = qp + (size_t)bh * N_ * HDP_;
  const __bf16* kb = kp + (size_t)bh * N_ * HDP_;
  const __bf16* vb = vT + (size_t)bh * HDP_ * N_;
  __bf16* pl = &sP[wave][0];

  v16bf qa[3];
#pragma unroll
  for (int j = 0; j < 3; ++j) qa[j] = load_frag(qb, HDP_, qt * 16, j * 32, lane);

  v8f accv[6];
#pragma unroll
  for (int t = 0; t < 6; ++t) accv[t] = zero8();
  float mst[8], lst[8];
#pragma unroll
  for (int i = 0; i < 8; ++i) { mst[i] = -1e30f; lst[i] = 0.0f; }

  const float scale = 0.11785113019775793f;    // 1/sqrt(72)
  const int col = lane & 15, half = lane >> 4;

  for (int kc = 0; kc < N_; kc += 32) {
    v8f s0 = zero8(), s1 = zero8();
#pragma unroll
    for (int j = 0; j < 3; ++j) {
      v16bf kf = load_frag(kb, HDP_, kc, j * 32, lane);
      s0 = wmma_bf16(qa[j], kf, s0);
    }
#pragma unroll
    for (int j = 0; j < 3; ++j) {
      v16bf kf = load_frag(kb, HDP_, kc + 16, j * 32, lane);
      s1 = wmma_bf16(qa[j], kf, s1);
    }
#pragma unroll
    for (int i = 0; i < 8; ++i) {
      float a0 = s0[i] * scale, a1 = s1[i] * scale;
      float r = rowmax16(fmaxf(a0, a1));       // DPP16 butterfly, no LDS traffic
      float mn = fmaxf(mst[i], r);
      float al = __expf(mst[i] - mn);
      float p0 = __expf(a0 - mn), p1 = __expf(a1 - mn);
      float rs = rowsum16(p0 + p1);
      lst[i] = lst[i] * al + rs;
      mst[i] = mn;
#pragma unroll
      for (int t = 0; t < 6; ++t) accv[t][i] *= al;
      const int m = i + (half << 3);
      pl[m * 32 + col]      = (__bf16)p0;
      pl[m * 32 + 16 + col] = (__bf16)p1;
    }
    asm volatile("s_wait_dscnt 0" ::: "memory");   // own-wave LDS P tile ready
    v16bf pa = load_frag(pl, 32, 0, 0, lane);
#pragma unroll
    for (int t = 0; t < 6; ++t) {
      v16bf vf = load_frag(vb, N_, t * 16, kc, lane);
      accv[t] = wmma_bf16(pa, vf, accv[t]);
    }
  }

#pragma unroll
  for (int t = 0; t < 6; ++t) {
#pragma unroll
    for (int i = 0; i < 8; ++i) {
      const int d = t * 16 + col;
      if (d < HD_) {
        const int tok = qt * 16 + i + (half << 3);
        o[((size_t)(b * N_ + tok)) * H_ + h * HD_ + d] = (__bf16)(accv[t][i] / lst[i]);
      }
    }
  }
}

// ---------------------------------------------------------------------- launch
extern "C" void kernel_launch(void* const* d_in, const int* in_sizes, int n_in,
                              void* d_out, int out_size, void* d_ws, size_t ws_size,
                              hipStream_t stream) {
  const float* x      = (const float*)d_in[0];
  const float* c      = (const float*)d_in[1];
  const float* w_mod  = (const float*)d_in[2];
  const float* b_mod  = (const float*)d_in[3];
  const float* w_qkv  = (const float*)d_in[4];
  const float* b_qkv  = (const float*)d_in[5];
  const float* g_q    = (const float*)d_in[6];
  const float* g_k    = (const float*)d_in[7];
  const float* w_proj = (const float*)d_in[8];
  const float* b_proj = (const float*)d_in[9];
  const float* w1     = (const float*)d_in[10];
  const float* b1     = (const float*)d_in[11];
  const float* w2     = (const float*)d_in[12];
  const float* b2     = (const float*)d_in[13];
  float* out = (float*)d_out;

  uint8_t* ws = (uint8_t*)d_ws;
  size_t cur = 0;
  auto alloc = [&](size_t bytes) -> void* {
    void* p = ws + cur;
    cur += (bytes + 255) & ~(size_t)255;
    return p;
  };
  float*  s_sc   = (float*)alloc((size_t)B_ * H_ * 4);
  float*  modb   = (float*)alloc((size_t)B_ * MOD6_ * 4);
  __bf16* wqkvT  = (__bf16*)alloc((size_t)3 * H_ * H_ * 2);
  __bf16* wprojT = (__bf16*)alloc((size_t)H_ * H_ * 2);
  __bf16* w1T    = (__bf16*)alloc((size_t)MLP_ * H_ * 2);
  __bf16* w2T    = (__bf16*)alloc((size_t)H_ * MLP_ * 2);
  __bf16* xn     = (__bf16*)alloc((size_t)ROWS_ * H_ * 2);
  __bf16* qkv    = (__bf16*)alloc((size_t)ROWS_ * 3 * H_ * 2);
  __bf16* qp     = (__bf16*)alloc((size_t)B_ * NH_ * N_ * HDP_ * 2);
  __bf16* kp     = (__bf16*)alloc((size_t)B_ * NH_ * N_ * HDP_ * 2);
  __bf16* vT     = (__bf16*)alloc((size_t)B_ * NH_ * HDP_ * N_ * 2);
  __bf16* ob     = (__bf16*)alloc((size_t)ROWS_ * H_ * 2);
  float*  x1     = (float*)alloc((size_t)ROWS_ * H_ * 4);
  __bf16* xn2    = (__bf16*)alloc((size_t)ROWS_ * H_ * 2);
  __bf16* hb     = (__bf16*)alloc((size_t)ROWS_ * MLP_ * 2);
  (void)ws_size; (void)in_sizes; (void)n_in; (void)out_size;

  auto cdiv = [](int a, int b) { return (a + b - 1) / b; };

  // conditioning path
  swish_kernel<<<cdiv(B_ * H_, 256), 256, 0, stream>>>(c, s_sc, B_ * H_);
  mod_gemm_kernel<<<cdiv(B_ * MOD6_, 256), 256, 0, stream>>>(s_sc, w_mod, b_mod, modb);

  // weights -> bf16 [N][K]
  convT_kernel<<<cdiv(H_ * 3 * H_, 256), 256, 0, stream>>>(w_qkv, wqkvT, H_, 3 * H_);
  convT_kernel<<<cdiv(H_ * H_, 256), 256, 0, stream>>>(w_proj, wprojT, H_, H_);
  convT_kernel<<<cdiv(H_ * MLP_, 256), 256, 0, stream>>>(w1, w1T, H_, MLP_);
  convT_kernel<<<cdiv(MLP_ * H_, 256), 256, 0, stream>>>(w2, w2T, MLP_, H_);

  // attention branch
  ln_mod_kernel<<<ROWS_, 256, 0, stream>>>(x, modb, 0, H_, xn);
  gemm_kernel<0><<<(ROWS_ / BM_) * ((3 * H_) / 64), 256, 0, stream>>>(
      xn, wqkvT, b_qkv, nullptr, qkv, nullptr, nullptr, ROWS_, 3 * H_, H_);
  qk_ln_pack_kernel<<<cdiv(B_ * NH_ * N_, 256), 256, 0, stream>>>(qkv, g_q, g_k, qp, kp, vT);
  attn_kernel<<<cdiv(B_ * NH_ * (N_ / 16), 8), 256, 0, stream>>>(qp, kp, vT, ob);
  gemm_kernel<2><<<(ROWS_ / BM_) * (H_ / 64), 256, 0, stream>>>(
      ob, wprojT, b_proj, x1, nullptr, x, modb + 2 * H_, ROWS_, H_, H_);

  // MLP branch
  ln_mod_kernel<<<ROWS_, 256, 0, stream>>>(x1, modb, 3 * H_, 4 * H_, xn2);
  gemm_kernel<1><<<(ROWS_ / BM_) * (MLP_ / 64), 256, 0, stream>>>(
      xn2, w1T, b1, nullptr, hb, nullptr, nullptr, ROWS_, MLP_, H_);
  gemm_kernel<2><<<(ROWS_ / BM_) * (H_ / 64), 256, 0, stream>>>(
      hb, w2T, b2, out, nullptr, x1, modb + 5 * H_, ROWS_, H_, MLP_);
}
